// MultiHeadAttention_48284022341836
// MI455X (gfx1250) — matrix operand order repaired and checked by reading, not compiled, not run on hardware
//
#include <hip/hip_runtime.h>
#include <stdint.h>

// ---------------------------------------------------------------------------
// MHA forward for gfx1250 (MI455X): bf16 WMMA, fp32 accumulate, async-to-LDS
// tile staging (ASYNCcnt path) + global prefetch. B=4, T=2048, D=2048, H=16.
// ---------------------------------------------------------------------------

typedef __attribute__((ext_vector_type(16))) __bf16        v16bf;
typedef __attribute__((ext_vector_type(8)))  float         v8f;
typedef __attribute__((ext_vector_type(4)))  unsigned int  v4u;
typedef __attribute__((ext_vector_type(4)))  int           v4i;

#define DEV __device__ __forceinline__

#if defined(__has_builtin)
#if __has_builtin(__builtin_amdgcn_global_load_async_to_lds_b128) && \
    __has_builtin(__builtin_amdgcn_s_wait_asynccnt)
#define HAVE_ASYNC 1
#endif
#endif
#ifndef HAVE_ASYNC
#define HAVE_ASYNC 0
#endif

DEV unsigned short f32_to_bf16(float f) {
  union { float f; unsigned int u; } c; c.f = f;
  unsigned int u = c.u;
  u += 0x7FFFu + ((u >> 16) & 1u);          // round to nearest even
  return (unsigned short)(u >> 16);
}

DEV v16bf make_frag(v4u lo, v4u hi) {
  union { v4u u[2]; v16bf v; } c;
  c.u[0] = lo; c.u[1] = hi;
  return c.v;
}

DEV v8f wmma_bf16(v16bf a, v16bf b, v8f c) {
  return __builtin_amdgcn_wmma_f32_16x16x32_bf16(false, a, false, b, (short)0, c,
                                                 false, false);
}

// 16B global -> LDS, async (ASYNCcnt) when available, else through VGPRs.
DEV void stage_b128(const unsigned short* g, unsigned short* l) {
#if HAVE_ASYNC
  __builtin_amdgcn_global_load_async_to_lds_b128(
      (__attribute__((address_space(1))) v4i*)g,
      (__attribute__((address_space(3))) v4i*)l, 0, 0);
#else
  *reinterpret_cast<v4u*>(l) = *reinterpret_cast<const v4u*>(g);
#endif
}

DEV void wait_stage() {
#if HAVE_ASYNC
  __builtin_amdgcn_s_wait_asynccnt(0);
#endif
}

// ---------------------------------------------------------------------------
// fp32 -> bf16 elementwise conversion (4 elems / thread)
// ---------------------------------------------------------------------------
__global__ __launch_bounds__(256) void cvt_f32_bf16(const float* __restrict__ in,
                                                    unsigned short* __restrict__ out,
                                                    int n4) {
  int i = blockIdx.x * blockDim.x + threadIdx.x;
  if (i < n4) {
    float4 f = reinterpret_cast<const float4*>(in)[i];
    union { unsigned short h[4]; uint2 u; } o;
    o.h[0] = f32_to_bf16(f.x);
    o.h[1] = f32_to_bf16(f.y);
    o.h[2] = f32_to_bf16(f.z);
    o.h[3] = f32_to_bf16(f.w);
    reinterpret_cast<uint2*>(out)[i] = o.u;
  }
}

// ---------------------------------------------------------------------------
// C[M=8192, N=2048] = A[M,K=2048] @ W[N,K]^T + bias   (bf16 in, f32 accum)
// mode 0: write bf16 head-split [B,H,T,HD];  mode 1: write f32 row-major.
// Block: 256 threads = 8 waves; tile 256x128; K-step 32, double-buffered LDS.
// Wave grid 4(m) x 2(n): each wave 64x64 = 4x4 WMMA fragments (16 wmma/step).
// ---------------------------------------------------------------------------
__global__ __launch_bounds__(256) void gemm_bf16(
    const unsigned short* __restrict__ A,
    const unsigned short* __restrict__ W,
    const float* __restrict__ bias,
    unsigned short* __restrict__ outH,
    float* __restrict__ outF,
    int mode) {
  const int K = 2048;
  const int NKT = K / 32;
  __shared__ unsigned short Xs[2][256][32];
  __shared__ unsigned short Wsh[2][128][32];

  const int tid  = threadIdx.x;
  const int lane = tid & 31;
  const int w    = tid >> 5;
  const int wm   = w & 3;          // 4 m-slots * 64 rows
  const int wn   = w >> 2;         // 2 n-slots * 64 cols
  const int m0   = blockIdx.y * 256;
  const int n0   = blockIdx.x * 128;
  const int ln   = lane & 15;
  const int half = lane >> 4;

  const v8f vzero = {0.f,0.f,0.f,0.f,0.f,0.f,0.f,0.f};
  v8f acc[4][4];
#pragma unroll
  for (int i = 0; i < 4; ++i)
#pragma unroll
    for (int j = 0; j < 4; ++j) acc[i][j] = vzero;

  // Cooperative staging: 1024 b128 (X) + 512 b128 (W) per tile.
  auto stage = [&](int buf, int k0) {
    int idx = tid;
#pragma unroll
    for (int r = 0; r < 4; ++r, idx += 256) {
      int row = idx >> 2, c = idx & 3;
      stage_b128(A + (size_t)(m0 + row) * K + k0 + c * 8, &Xs[buf][row][c * 8]);
    }
    idx = tid;
#pragma unroll
    for (int r = 0; r < 2; ++r, idx += 256) {
      int row = idx >> 2, c = idx & 3;
      stage_b128(W + (size_t)(n0 + row) * K + k0 + c * 8, &Wsh[buf][row][c * 8]);
    }
  };

  stage(0, 0);
  wait_stage();
  __syncthreads();

  for (int kt = 0; kt < NKT; ++kt) {
    const int cur = kt & 1;
    if (kt + 1 < NKT) stage(cur ^ 1, (kt + 1) * 32);
    if (kt + 2 < NKT) {
      __builtin_prefetch(A + (size_t)(m0 + tid) * K + (kt + 2) * 32, 0, 3);
      if (tid < 128)
        __builtin_prefetch(W + (size_t)(n0 + tid) * K + (kt + 2) * 32, 0, 3);
    }

    v16bf afrag[4], bfrag[4];
#pragma unroll
    for (int mi = 0; mi < 4; ++mi) {
      const unsigned short* rp = &Xs[cur][wm * 64 + mi * 16 + ln][0];
      afrag[mi] = make_frag(*reinterpret_cast<const v4u*>(rp + 8 * half),
                            *reinterpret_cast<const v4u*>(rp + 16 + 8 * half));
    }
#pragma unroll
    for (int ni = 0; ni < 4; ++ni) {
      const unsigned short* rp = &Wsh[cur][wn * 64 + ni * 16 + ln][0];
      bfrag[ni] = make_frag(*reinterpret_cast<const v4u*>(rp + 8 * half),
                            *reinterpret_cast<const v4u*>(rp + 16 + 8 * half));
    }
#pragma unroll
    for (int mi = 0; mi < 4; ++mi)
#pragma unroll
      for (int ni = 0; ni < 4; ++ni)
        acc[mi][ni] = wmma_bf16(afrag[mi], bfrag[ni], acc[mi][ni]);

    wait_stage();
    __syncthreads();
  }

  // epilogue: bias + store
#pragma unroll
  for (int mi = 0; mi < 4; ++mi) {
#pragma unroll
    for (int ni = 0; ni < 4; ++ni) {
      int n = n0 + wn * 64 + ni * 16 + ln;
      float bv = bias[n];
      int mrow = m0 + wm * 64 + mi * 16 + half * 8;
      if (mode == 0) {
        int h = n >> 7, dd = n & 127;
#pragma unroll
        for (int j = 0; j < 8; ++j) {
          int m = mrow + j;
          int bb = m >> 11, t = m & 2047;
          outH[(((size_t)(bb * 16 + h) * 2048 + t) << 7) + dd] =
              f32_to_bf16(acc[mi][ni][j] + bv);
        }
      } else {
#pragma unroll
        for (int j = 0; j < 8; ++j) {
          int m = mrow + j;
          outF[(size_t)m * 2048 + n] = acc[mi][ni][j] + bv;
        }
      }
    }
  }
}

// ---------------------------------------------------------------------------
// Flash attention: one block = (b,h) x 128 queries; 8 waves x 16 queries.
// K staged async into LDS; V staged transposed; online softmax; WMMA matmuls.
// ---------------------------------------------------------------------------
__global__ __launch_bounds__(256) void attn_fwd(
    const unsigned short* __restrict__ Q,
    const unsigned short* __restrict__ Kg,
    const unsigned short* __restrict__ Vg,
    unsigned short* __restrict__ ctx) {
  const int T = 2048, HD = 128;
  __shared__ unsigned short Kl[32][128];       // [key][d]
  __shared__ unsigned short Vt[128][32];       // [d][key]  (transposed)
  __shared__ unsigned short Pw[8][16][32];     // per-wave P staging

  const int tid  = threadIdx.x;
  const int lane = tid & 31;
  const int w    = tid >> 5;
  const int ln   = lane & 15;
  const int half = lane >> 4;
  const int bh   = blockIdx.y;                 // b*16 + h
  const int q0   = blockIdx.x * 128;
  const int qbase = q0 + w * 16;

  const size_t headoff = (size_t)bh * T * HD;
  const unsigned short* Qh = Q + headoff;
  const unsigned short* Kh = Kg + headoff;
  const unsigned short* Vh = Vg + headoff;

  v16bf qf[4];
  {
    const unsigned short* qr = Qh + (size_t)(qbase + ln) * HD;
#pragma unroll
    for (int ds = 0; ds < 4; ++ds) {
      v4u lo = *reinterpret_cast<const v4u*>(qr + ds * 32 + 8 * half);
      v4u hi = *reinterpret_cast<const v4u*>(qr + ds * 32 + 16 + 8 * half);
      qf[ds] = make_frag(lo, hi);
    }
  }

  const v8f vzero = {0.f,0.f,0.f,0.f,0.f,0.f,0.f,0.f};
  v8f o[8];
#pragma unroll
  for (int i = 0; i < 8; ++i) o[i] = vzero;
  float mrow[8], lrow[8];
#pragma unroll
  for (int j = 0; j < 8; ++j) { mrow[j] = -1e30f; lrow[j] = 0.f; }

  const float ssc = 0.08838834764831845f * 1.44269504088896340736f;
  const int nkb = ((q0 + 127) >> 5) + 1;       // causal: keys [0, q0+128)

  for (int kb = 0; kb < nkb; ++kb) {
    const int kb0 = kb * 32;
    // ---- stage K (async) and V (transposed) tiles ----
    {
      int idx = tid;
#pragma unroll
      for (int r = 0; r < 2; ++r, idx += 256) {
        int row = idx >> 4, c = idx & 15;
        stage_b128(Kh + (size_t)(kb0 + row) * HD + c * 8, &Kl[row][c * 8]);
      }
      int kk = tid >> 3, dg = (tid & 7) * 16;
      const unsigned short* vr = Vh + (size_t)(kb0 + kk) * HD + dg;
#pragma unroll
      for (int e = 0; e < 16; ++e) Vt[dg + e][kk] = vr[e];
      if (kb + 1 < nkb) {
        __builtin_prefetch(Kh + (size_t)(kb0 + 32 + (tid >> 3)) * HD, 0, 3);
        __builtin_prefetch(Vh + (size_t)(kb0 + 32 + (tid >> 3)) * HD, 0, 3);
      }
    }
    wait_stage();
    __syncthreads();

    // ---- scores: 2 key tiles x 4 k-steps ----
    v8f s[2];
    s[0] = vzero; s[1] = vzero;
#pragma unroll
    for (int nt = 0; nt < 2; ++nt) {
#pragma unroll
      for (int ds = 0; ds < 4; ++ds) {
        const unsigned short* rp = &Kl[nt * 16 + ln][ds * 32];
        v4u lo = *reinterpret_cast<const v4u*>(rp + 8 * half);
        v4u hi = *reinterpret_cast<const v4u*>(rp + 16 + 8 * half);
        s[nt] = wmma_bf16(qf[ds], make_frag(lo, hi), s[nt]);
      }
    }

    // ---- causal mask + online softmax ----
#pragma unroll
    for (int j = 0; j < 8; ++j) {
      int q = qbase + half * 8 + j;
#pragma unroll
      for (int nt = 0; nt < 2; ++nt) {
        int key = kb0 + nt * 16 + ln;
        if (key > q) s[nt][j] = -1e30f;
      }
      float v = fmaxf(s[0][j], s[1][j]);
      v = fmaxf(v, __shfl_xor(v, 1));
      v = fmaxf(v, __shfl_xor(v, 2));
      v = fmaxf(v, __shfl_xor(v, 4));
      v = fmaxf(v, __shfl_xor(v, 8));
      float mn = fmaxf(mrow[j], v);
      float alpha = __builtin_amdgcn_exp2f((mrow[j] - mn) * ssc);
      mrow[j] = mn;
      float rs = 0.f;
#pragma unroll
      for (int nt = 0; nt < 2; ++nt) {
        float p = __builtin_amdgcn_exp2f((s[nt][j] - mn) * ssc);
        s[nt][j] = p;
        rs += p;
      }
      rs += __shfl_xor(rs, 1);
      rs += __shfl_xor(rs, 2);
      rs += __shfl_xor(rs, 4);
      rs += __shfl_xor(rs, 8);
      lrow[j] = lrow[j] * alpha + rs;
#pragma unroll
      for (int dt = 0; dt < 8; ++dt) o[dt][j] *= alpha;
    }

    // ---- C-layout -> A-layout for P via per-wave LDS (in-order DS) ----
#pragma unroll
    for (int j = 0; j < 8; ++j) {
      Pw[w][half * 8 + j][ln]      = f32_to_bf16(s[0][j]);
      Pw[w][half * 8 + j][16 + ln] = f32_to_bf16(s[1][j]);
    }
    asm volatile("" ::: "memory");
    v16bf pf;
    {
      const unsigned short* rp = &Pw[w][ln][0];
      v4u lo = *reinterpret_cast<const v4u*>(rp + 8 * half);
      v4u hi = *reinterpret_cast<const v4u*>(rp + 16 + 8 * half);
      pf = make_frag(lo, hi);
    }

    // ---- O += P @ V : 8 d-tiles ----
#pragma unroll
    for (int dt = 0; dt < 8; ++dt) {
      const unsigned short* rp = &Vt[dt * 16 + ln][0];
      v4u lo = *reinterpret_cast<const v4u*>(rp + 8 * half);
      v4u hi = *reinterpret_cast<const v4u*>(rp + 16 + 8 * half);
      o[dt] = wmma_bf16(pf, make_frag(lo, hi), o[dt]);
    }
    __syncthreads();
  }

  // ---- normalize and store ctx bf16 [b*T+t][h*128+d] ----
  const int bb = bh >> 4, h = bh & 15;
#pragma unroll
  for (int j = 0; j < 8; ++j) {
    float inv = 1.0f / lrow[j];
    int t = qbase + half * 8 + j;
    size_t rowoff = (((size_t)(bb * 2048 + t)) << 11) + (h << 7) + ln;
#pragma unroll
    for (int dt = 0; dt < 8; ++dt)
      ctx[rowoff + (size_t)dt * 16] = f32_to_bf16(o[dt][j] * inv);
  }
}

// ---------------------------------------------------------------------------
extern "C" void kernel_launch(void* const* d_in, const int* in_sizes, int n_in,
                              void* d_out, int out_size, void* d_ws, size_t ws_size,
                              hipStream_t stream) {
  (void)in_sizes; (void)n_in; (void)out_size; (void)ws_size;
  const float* x  = (const float*)d_in[0];
  const float* Wq = (const float*)d_in[1];
  const float* bq = (const float*)d_in[2];
  const float* Wk = (const float*)d_in[3];
  const float* bk = (const float*)d_in[4];
  const float* Wv = (const float*)d_in[5];
  const float* bv = (const float*)d_in[6];
  const float* Wo = (const float*)d_in[7];
  const float* bo = (const float*)d_in[8];
  float* out = (float*)d_out;

  const size_t NX = (size_t)8192 * 2048;
  const size_t NW = (size_t)2048 * 2048;

  unsigned short* ws  = (unsigned short*)d_ws;
  unsigned short* xb  = ws;  ws += NX;
  unsigned short* wqb = ws;  ws += NW;
  unsigned short* wkb = ws;  ws += NW;
  unsigned short* wvb = ws;  ws += NW;
  unsigned short* wob = ws;  ws += NW;
  unsigned short* qb  = ws;  ws += NX;
  unsigned short* kbf = ws;  ws += NX;
  unsigned short* vbf = ws;  ws += NX;
  unsigned short* ctxb = xb;               // reuse: x dead after V projection

  cvt_f32_bf16<<<dim3((unsigned)(NX / 4 / 256)), 256, 0, stream>>>(x,  xb,  (int)(NX / 4));
  cvt_f32_bf16<<<dim3((unsigned)(NW / 4 / 256)), 256, 0, stream>>>(Wq, wqb, (int)(NW / 4));
  cvt_f32_bf16<<<dim3((unsigned)(NW / 4 / 256)), 256, 0, stream>>>(Wk, wkb, (int)(NW / 4));
  cvt_f32_bf16<<<dim3((unsigned)(NW / 4 / 256)), 256, 0, stream>>>(Wv, wvb, (int)(NW / 4));
  cvt_f32_bf16<<<dim3((unsigned)(NW / 4 / 256)), 256, 0, stream>>>(Wo, wob, (int)(NW / 4));

  dim3 gg(16, 32), blk(256);   // N/128=16, M/256=32
  gemm_bf16<<<gg, blk, 0, stream>>>(xb, wqb, bq, qb,  nullptr, 0);
  gemm_bf16<<<gg, blk, 0, stream>>>(xb, wkb, bk, kbf, nullptr, 0);
  gemm_bf16<<<gg, blk, 0, stream>>>(xb, wvb, bv, vbf, nullptr, 0);

  attn_fwd<<<dim3(16, 64), 256, 0, stream>>>(qb, kbf, vbf, ctxb);

  gemm_bf16<<<gg, blk, 0, stream>>>(ctxb, wob, bo, nullptr, out, 1);
}
